// Mlp_84275848282705
// MI455X (gfx1250) — 8-bit matrix operand order repaired and checked by reading, compiled, not run on hardware
//
#include <hip/hip_runtime.h>
#include <stdint.h>

typedef int v8i __attribute__((ext_vector_type(8)));

#define M_TOK 12608
#define C_DIM 768
#define H_DIM 3072
#define EPSQ  1e-8f

__device__ __forceinline__ unsigned fkey(float f) {
  unsigned u = __float_as_uint(f);
  return (u & 0x80000000u) ? ~u : (u | 0x80000000u);
}
__device__ __forceinline__ float fdec(unsigned k) {
  return __uint_as_float((k & 0x80000000u) ? (k & 0x7FFFFFFFu) : ~k);
}
__device__ __forceinline__ float gelu_exact(float v) {
  return 0.5f * v * (1.0f + erff(v * 0.70710678118654752f));
}

__global__ void k_init(unsigned* scal, unsigned* xchmax, unsigned* wcolmax) {
  int t = threadIdx.x;
  for (int i = t; i < C_DIM; i += 256) { xchmax[i] = 0u; wcolmax[i] = 0u; }
  if (t == 0) {
    scal[0] = 0xFFFFFFFFu; scal[1] = 0u; scal[2] = 0u;
    scal[3] = 0xFFFFFFFFu; scal[4] = 0u;
  }
}

__global__ void k_absmax_x(const float* __restrict__ x, unsigned* __restrict__ xchmax) {
  int r0 = blockIdx.x * 64, t = threadIdx.x;
#pragma unroll
  for (int j = 0; j < 3; ++j) {
    int c = t + j * 256;
    float m = 0.f;
    for (int r = 0; r < 64; ++r)
      m = fmaxf(m, fabsf(x[(size_t)(r0 + r) * C_DIM + c]));
    atomicMax(&xchmax[c], __float_as_uint(m));
  }
}

__global__ void k_absmax_w1col(const float* __restrict__ w1, unsigned* __restrict__ wcolmax) {
  int r0 = blockIdx.x * 64, t = threadIdx.x;
#pragma unroll
  for (int j = 0; j < 3; ++j) {
    int c = t + j * 256;
    float m = 0.f;
    for (int r = 0; r < 64; ++r)
      m = fmaxf(m, fabsf(w1[(size_t)(r0 + r) * C_DIM + c]));
    atomicMax(&wcolmax[c], __float_as_uint(m));
  }
}

__global__ void k_scales(const unsigned* __restrict__ xchmax,
                         const unsigned* __restrict__ wcolmax,
                         float* __restrict__ cs) {
  int c = blockIdx.x * 256 + threadIdx.x;
  if (c >= C_DIM) return;
  float gm = __uint_as_float(xchmax[c]);
  float wm = fmaxf(__uint_as_float(wcolmax[c]), 1e-20f);
  float v = sqrtf(gm) / sqrtf(wm);
  float out = 1.f;
  if (v > 0.f && v < 3.0e38f) {
    float y = floorf(log2f(v));
    float p = exp2f(y);
    out = ((v - p) > (2.f * p - v)) ? (2.f * p) : p;
  }
  cs[c] = out;
}

__global__ void k_minmax_xs(const float* __restrict__ x, const float* __restrict__ cs,
                            unsigned* __restrict__ scal) {
  __shared__ float rmin[256], rmax[256];
  float lmin = 3.4e38f, lmax = -3.4e38f;
  size_t total = (size_t)M_TOK * C_DIM;
  for (size_t i = (size_t)blockIdx.x * 256 + threadIdx.x; i < total; i += (size_t)gridDim.x * 256) {
    float v = x[i] / cs[i % C_DIM];
    lmin = fminf(lmin, v); lmax = fmaxf(lmax, v);
  }
  int t = threadIdx.x;
  rmin[t] = lmin; rmax[t] = lmax; __syncthreads();
  for (int s = 128; s > 0; s >>= 1) {
    if (t < s) { rmin[t] = fminf(rmin[t], rmin[t + s]); rmax[t] = fmaxf(rmax[t], rmax[t + s]); }
    __syncthreads();
  }
  if (t == 0) { atomicMin(&scal[0], fkey(rmin[0])); atomicMax(&scal[1], fkey(rmax[0])); }
}

__global__ void k_quant_x(const float* __restrict__ x, const float* __restrict__ cs,
                          const unsigned* __restrict__ scal, uint8_t* __restrict__ aq) {
  float xmin = fminf(fdec(scal[0]), 0.f);
  float xmax = fmaxf(fdec(scal[1]), 0.f);
  float s = fmaxf((xmax - xmin) * (1.f / 255.f), EPSQ);
  float zp = rintf(-xmin / s);
  size_t total = (size_t)M_TOK * C_DIM;
  for (size_t i = (size_t)blockIdx.x * 256 + threadIdx.x; i < total; i += (size_t)gridDim.x * 256) {
    float v = x[i] / cs[i % C_DIM];
    float q = fminf(fmaxf(rintf(v / s) + zp, 0.f), 255.f);
    aq[i] = (uint8_t)q;
  }
}

__global__ void k_quant_w1(const float* __restrict__ w1, const float* __restrict__ cs,
                           int8_t* __restrict__ w1q, float* __restrict__ wscale,
                           int* __restrict__ colsum) {
  __shared__ float rmax[256];
  __shared__ int rsum[256];
  int h = blockIdx.x, t = threadIdx.x;
  float v[3]; float m = 0.f;
#pragma unroll
  for (int j = 0; j < 3; ++j) {
    int c = t + j * 256;
    v[j] = w1[(size_t)h * C_DIM + c] * cs[c];
    m = fmaxf(m, fabsf(v[j]));
  }
  rmax[t] = m; __syncthreads();
  for (int s = 128; s > 0; s >>= 1) { if (t < s) rmax[t] = fmaxf(rmax[t], rmax[t + s]); __syncthreads(); }
  float sc = fmaxf(rmax[0] * (1.f / 127.f), EPSQ);
  int ssum = 0;
#pragma unroll
  for (int j = 0; j < 3; ++j) {
    int c = t + j * 256;
    int qi = (int)fminf(fmaxf(rintf(v[j] / sc), -128.f), 127.f);
    w1q[(size_t)h * C_DIM + c] = (int8_t)qi;
    ssum += qi;
  }
  rsum[t] = ssum; __syncthreads();
  for (int s = 128; s > 0; s >>= 1) { if (t < s) rsum[t] += rsum[t + s]; __syncthreads(); }
  if (t == 0) { wscale[h] = sc; colsum[h] = rsum[0]; }
}

__global__ void k_quant_w2(const float* __restrict__ w2, int8_t* __restrict__ w2q,
                           float* __restrict__ wscale) {
  __shared__ float rmax[256];
  int c = blockIdx.x, t = threadIdx.x;
  float v[12]; float m = 0.f;
#pragma unroll
  for (int j = 0; j < 12; ++j) {
    v[j] = w2[(size_t)c * H_DIM + t + j * 256];
    m = fmaxf(m, fabsf(v[j]));
  }
  rmax[t] = m; __syncthreads();
  for (int s = 128; s > 0; s >>= 1) { if (t < s) rmax[t] = fmaxf(rmax[t], rmax[t + s]); __syncthreads(); }
  float sc = fmaxf(rmax[0] * (1.f / 127.f), EPSQ);
#pragma unroll
  for (int j = 0; j < 12; ++j) {
    int qi = (int)fminf(fmaxf(rintf(v[j] / sc), -128.f), 127.f);
    w2q[(size_t)c * H_DIM + t + j * 256] = (int8_t)qi;
  }
  if (t == 0) wscale[c] = sc;
}

__global__ __launch_bounds__(256) void k_gemm1(
    const uint8_t* __restrict__ Aq, const int8_t* __restrict__ Wq,
    const int* __restrict__ colsum, const float* __restrict__ wsc,
    const float* __restrict__ bias, const unsigned* __restrict__ scal,
    _Float16* __restrict__ Hout, unsigned* __restrict__ habs) {
  const int Mm = M_TOK, Nn = H_DIM, Kk = C_DIM;
  int tid = threadIdx.x;
  int wave = tid >> 5, lane = tid & 31;
  int wm = wave >> 1, wn = wave & 1;
  int half = lane >> 4, l15 = lane & 15;
  int m_base = blockIdx.x * 128 + wm * 32;
  int n_base = blockIdx.y * 64 + wn * 32;

  float xmin = fminf(fdec(scal[0]), 0.f);
  float xmax = fmaxf(fdec(scal[1]), 0.f);
  float sx = fmaxf((xmax - xmin) * (1.f / 255.f), EPSQ);
  float zp = rintf(-xmin / sx);

  int mrow[2];
  mrow[0] = m_base + l15;      if (mrow[0] > Mm - 1) mrow[0] = Mm - 1;
  mrow[1] = m_base + 16 + l15; if (mrow[1] > Mm - 1) mrow[1] = Mm - 1;

  v8i acc[2][2] = {};

  for (int k0 = 0; k0 < Kk; k0 += 64) {
    v8i a[2], b[2];
#pragma unroll
    for (int s = 0; s < 2; ++s) {
      const uint8_t* ap = Aq + (size_t)mrow[s] * Kk + k0 + half * 16;
#pragma unroll
      for (int j = 0; j < 4; ++j) {
        uint2 d = *(const uint2*)(ap + (j & 1) * 8 + (j >> 1) * 32);
        a[s][2 * j] = (int)d.x; a[s][2 * j + 1] = (int)d.y;
      }
      int col = n_base + s * 16 + l15;
      const int8_t* bp = Wq + (size_t)col * Kk + k0 + half * 16;
      int4 q0 = *(const int4*)(bp);
      int4 q1 = *(const int4*)(bp + 32);
      b[s][0] = q0.x; b[s][1] = q0.y; b[s][2] = q0.z; b[s][3] = q0.w;
      b[s][4] = q1.x; b[s][5] = q1.y; b[s][6] = q1.z; b[s][7] = q1.w;
    }
    acc[0][0] = __builtin_amdgcn_wmma_i32_16x16x64_iu8(false, a[0], true, b[0], acc[0][0], false, false);
    acc[0][1] = __builtin_amdgcn_wmma_i32_16x16x64_iu8(false, a[0], true, b[1], acc[0][1], false, false);
    acc[1][0] = __builtin_amdgcn_wmma_i32_16x16x64_iu8(false, a[1], true, b[0], acc[1][0], false, false);
    acc[1][1] = __builtin_amdgcn_wmma_i32_16x16x64_iu8(false, a[1], true, b[1], acc[1][1], false, false);
  }

  float lmax = 0.f;
#pragma unroll
  for (int sm = 0; sm < 2; ++sm) {
#pragma unroll
    for (int sn = 0; sn < 2; ++sn) {
      int n = n_base + sn * 16 + l15;
      float wsn = sx * wsc[n];
      float bn = bias[n];
      float csn = (float)colsum[n];
#pragma unroll
      for (int r = 0; r < 8; ++r) {
        int m = m_base + sm * 16 + half * 8 + r;
        if (m < Mm) {
          float v = ((float)acc[sm][sn][r] - zp * csn) * wsn + bn;
          float g = gelu_exact(v);
          Hout[(size_t)m * Nn + n] = (_Float16)g;
          lmax = fmaxf(lmax, fabsf(g));
        }
      }
    }
  }
  __shared__ float red[256];
  red[tid] = lmax; __syncthreads();
  for (int s = 128; s > 0; s >>= 1) { if (tid < s) red[tid] = fmaxf(red[tid], red[tid + s]); __syncthreads(); }
  if (tid == 0) atomicMax(habs, __float_as_uint(red[0]));
}

__global__ void k_quant_h(const _Float16* __restrict__ hbuf, const unsigned* __restrict__ scal,
                          int8_t* __restrict__ hq) {
  float sc = fmaxf(__uint_as_float(scal[2]) * (1.f / 127.f), EPSQ);
  size_t total = (size_t)M_TOK * H_DIM;
  for (size_t i = (size_t)blockIdx.x * 256 + threadIdx.x; i < total; i += (size_t)gridDim.x * 256) {
    float v = (float)hbuf[i];
    hq[i] = (int8_t)(int)fminf(fmaxf(rintf(v / sc), -128.f), 127.f);
  }
}

__global__ __launch_bounds__(256) void k_gemm2(
    const int8_t* __restrict__ Aq, const int8_t* __restrict__ Wq,
    const float* __restrict__ wsc, const float* __restrict__ bias,
    unsigned* __restrict__ scal, float* __restrict__ Out) {
  const int Mm = M_TOK, Nn = C_DIM, Kk = H_DIM;
  int tid = threadIdx.x;
  int wave = tid >> 5, lane = tid & 31;
  int wm = wave >> 1, wn = wave & 1;
  int half = lane >> 4, l15 = lane & 15;
  int m_base = blockIdx.x * 128 + wm * 32;
  int n_base = blockIdx.y * 64 + wn * 32;

  float sh = fmaxf(__uint_as_float(scal[2]) * (1.f / 127.f), EPSQ);

  int mrow[2];
  mrow[0] = m_base + l15;      if (mrow[0] > Mm - 1) mrow[0] = Mm - 1;
  mrow[1] = m_base + 16 + l15; if (mrow[1] > Mm - 1) mrow[1] = Mm - 1;

  v8i acc[2][2] = {};

  for (int k0 = 0; k0 < Kk; k0 += 64) {
    v8i a[2], b[2];
#pragma unroll
    for (int s = 0; s < 2; ++s) {
      const int8_t* ap = Aq + (size_t)mrow[s] * Kk + k0 + half * 16;
#pragma unroll
      for (int j = 0; j < 4; ++j) {
        uint2 d = *(const uint2*)(ap + (j & 1) * 8 + (j >> 1) * 32);
        a[s][2 * j] = (int)d.x; a[s][2 * j + 1] = (int)d.y;
      }
      int col = n_base + s * 16 + l15;
      const int8_t* bp = Wq + (size_t)col * Kk + k0 + half * 16;
      int4 q0 = *(const int4*)(bp);
      int4 q1 = *(const int4*)(bp + 32);
      b[s][0] = q0.x; b[s][1] = q0.y; b[s][2] = q0.z; b[s][3] = q0.w;
      b[s][4] = q1.x; b[s][5] = q1.y; b[s][6] = q1.z; b[s][7] = q1.w;
    }
    acc[0][0] = __builtin_amdgcn_wmma_i32_16x16x64_iu8(true, a[0], true, b[0], acc[0][0], false, false);
    acc[0][1] = __builtin_amdgcn_wmma_i32_16x16x64_iu8(true, a[0], true, b[1], acc[0][1], false, false);
    acc[1][0] = __builtin_amdgcn_wmma_i32_16x16x64_iu8(true, a[1], true, b[0], acc[1][0], false, false);
    acc[1][1] = __builtin_amdgcn_wmma_i32_16x16x64_iu8(true, a[1], true, b[1], acc[1][1], false, false);
  }

  float lmin = 3.4e38f, lmax = -3.4e38f;
#pragma unroll
  for (int sm = 0; sm < 2; ++sm) {
#pragma unroll
    for (int sn = 0; sn < 2; ++sn) {
      int n = n_base + sn * 16 + l15;
      float wsn = sh * wsc[n];
      float bn = bias[n];
#pragma unroll
      for (int r = 0; r < 8; ++r) {
        int m = m_base + sm * 16 + half * 8 + r;
        if (m < Mm) {
          float v = (float)acc[sm][sn][r] * wsn + bn;
          Out[(size_t)m * Nn + n] = v;
          lmin = fminf(lmin, v); lmax = fmaxf(lmax, v);
        }
      }
    }
  }
  __shared__ float rmin[256], rmax[256];
  rmin[tid] = lmin; rmax[tid] = lmax; __syncthreads();
  for (int s = 128; s > 0; s >>= 1) {
    if (tid < s) { rmin[tid] = fminf(rmin[tid], rmin[tid + s]); rmax[tid] = fmaxf(rmax[tid], rmax[tid + s]); }
    __syncthreads();
  }
  if (tid == 0) { atomicMin(&scal[3], fkey(rmin[0])); atomicMax(&scal[4], fkey(rmax[0])); }
}

__global__ void k_quant_out(float* __restrict__ out, const unsigned* __restrict__ scal) {
  float omin = fminf(fdec(scal[3]), 0.f);
  float omax = fmaxf(fdec(scal[4]), 0.f);
  float s = fmaxf((omax - omin) * (1.f / 255.f), EPSQ);
  float zp = rintf(-omin / s);
  size_t total = (size_t)M_TOK * C_DIM;
  for (size_t i = (size_t)blockIdx.x * 256 + threadIdx.x; i < total; i += (size_t)gridDim.x * 256) {
    float v = out[i];
    float q = fminf(fmaxf(rintf(v / s) + zp, 0.f), 255.f);
    out[i] = (q - zp) * s;
  }
}

extern "C" void kernel_launch(void* const* d_in, const int* in_sizes, int n_in,
                              void* d_out, int out_size, void* d_ws, size_t ws_size,
                              hipStream_t stream) {
  const float* x  = (const float*)d_in[0];
  const float* w1 = (const float*)d_in[1];
  const float* b1 = (const float*)d_in[2];
  const float* w2 = (const float*)d_in[3];
  const float* b2 = (const float*)d_in[4];
  float* out = (float*)d_out;

  char* ws = (char*)d_ws;
  size_t o = 0;
  auto take = [&](size_t b) { size_t r = o; o += (b + 255) & ~(size_t)255; return r; };
  unsigned* scal    = (unsigned*)(ws + take(64 * 4));
  unsigned* xchmax  = (unsigned*)(ws + take(C_DIM * 4));
  unsigned* wcolmax = (unsigned*)(ws + take(C_DIM * 4));
  float*    cs      = (float*)(ws + take(C_DIM * 4));
  float*    wscale1 = (float*)(ws + take(H_DIM * 4));
  int*      colsum1 = (int*)(ws + take(H_DIM * 4));
  float*    wscale2 = (float*)(ws + take(C_DIM * 4));
  uint8_t*  aq      = (uint8_t*)(ws + take((size_t)M_TOK * C_DIM));
  int8_t*   w1q     = (int8_t*)(ws + take((size_t)H_DIM * C_DIM));
  int8_t*   w2q     = (int8_t*)(ws + take((size_t)C_DIM * H_DIM));
  _Float16* hbuf    = (_Float16*)(ws + take((size_t)M_TOK * H_DIM * 2));
  int8_t*   hq      = (int8_t*)(ws + take((size_t)M_TOK * H_DIM));
  (void)ws_size; (void)in_sizes; (void)n_in; (void)out_size;

  k_init<<<1, 256, 0, stream>>>(scal, xchmax, wcolmax);
  k_absmax_x<<<M_TOK / 64, 256, 0, stream>>>(x, xchmax);
  k_absmax_w1col<<<H_DIM / 64, 256, 0, stream>>>(w1, wcolmax);
  k_scales<<<3, 256, 0, stream>>>(xchmax, wcolmax, cs);
  k_minmax_xs<<<1024, 256, 0, stream>>>(x, cs, scal);
  k_quant_x<<<2048, 256, 0, stream>>>(x, cs, scal, aq);
  k_quant_w1<<<H_DIM, 256, 0, stream>>>(w1, cs, w1q, wscale1, colsum1);
  k_quant_w2<<<C_DIM, 256, 0, stream>>>(w2, w2q, wscale2);

  dim3 g1((M_TOK + 127) / 128, H_DIM / 64);
  k_gemm1<<<g1, 256, 0, stream>>>(aq, w1q, colsum1, wscale1, b1, scal, hbuf, &scal[2]);
  k_quant_h<<<4096, 256, 0, stream>>>(hbuf, scal, hq);

  dim3 g2((M_TOK + 127) / 128, C_DIM / 64);
  k_gemm2<<<g2, 256, 0, stream>>>(hq, w2q, wscale2, b2, scal, out);
  k_quant_out<<<2048, 256, 0, stream>>>(out, scal);
}
